// NoisyTopkRouter_89498528514677
// MI455X (gfx1250) — matrix operand order repaired and checked by reading, not compiled, not run on hardware
//
#include <hip/hip_runtime.h>
#include <hip/hip_bf16.h>
#include <math.h>

typedef __attribute__((ext_vector_type(16))) __bf16 v16bf;
typedef __attribute__((ext_vector_type(8)))  __bf16 v8bf;
typedef __attribute__((ext_vector_type(4)))  __bf16 v4bf;
typedef __attribute__((ext_vector_type(8)))  float  v8f;
typedef __attribute__((ext_vector_type(4)))  float  v4f;
typedef int v4i __attribute__((vector_size(16)));

#define NTOK 32768   // B*T
#define CDIM 1024
#define HDIM 4096
#define EXP  8
#define TM 128
#define TN 128
#define TK 64
#define LDT 72       // padded LDS row stride (elements): 144B = 9*16B
#define TILE_ELT (TM * LDT)
#define BUF_ELT  (2 * TILE_ELT)   // A tile + B tile

#if defined(__gfx1250__) && __has_builtin(__builtin_amdgcn_global_load_async_to_lds_b128)
#define USE_ASYNC_LDS 1
#else
#define USE_ASYNC_LDS 0
#endif

__device__ __forceinline__ void async_wait0() {
#if USE_ASYNC_LDS
#if __has_builtin(__builtin_amdgcn_s_wait_asynccnt)
  __builtin_amdgcn_s_wait_asynccnt(0);
#else
  asm volatile("s_wait_asynccnt 0x0" ::: "memory");
#endif
#endif
}

// copy a 128-row x 64-col bf16 tile (global row stride gstride) into LDS (stride LDT)
__device__ __forceinline__ void stage_tile(const __bf16* __restrict__ g, size_t gstride,
                                           __bf16* lds, int tid) {
#pragma unroll
  for (int i = 0; i < 4; ++i) {
    int s = tid + i * 256;           // 1024 slots of 8 bf16
    int row = s >> 3, kc = s & 7;
    const __bf16* gp = g + (size_t)row * gstride + kc * 8;
    __bf16* lp = lds + row * LDT + kc * 8;
#if USE_ASYNC_LDS
    __builtin_amdgcn_global_load_async_to_lds_b128((v4i*)gp, (v4i*)lp, 0, 0);
#else
    *(v8bf*)lp = *(const v8bf*)gp;
#endif
  }
}

// build v16bf fragment from LDS (two 16B ds loads); layout per ISA 7.12.2
__device__ __forceinline__ v16bf frag16(const __bf16* p, int hi_off) {
  v8bf lo = *(const v8bf*)p;
  v8bf hi = *(const v8bf*)(p + hi_off);
  v16bf r;
#pragma unroll
  for (int j = 0; j < 8; ++j) { r[j] = lo[j]; r[8 + j] = hi[j]; }
  return r;
}

// ---------------- fp32 -> bf16 conversion (vector x4) ----------------
__global__ __launch_bounds__(256) void cvt_f32_bf16(const float* __restrict__ src,
                                                    __bf16* __restrict__ dst, int n4) {
  int i = blockIdx.x * blockDim.x + threadIdx.x;
  if (i < n4) {
    v4f v = *(const v4f*)(src + 4 * (size_t)i);
    v4bf o;
    o[0] = (__bf16)v[0]; o[1] = (__bf16)v[1]; o[2] = (__bf16)v[2]; o[3] = (__bf16)v[3];
    *(v4bf*)(dst + 4 * (size_t)i) = o;
  }
}

// ---------------- fp32 [K,N] -> bf16 transposed [N,K] (64x64 LDS tiles) ----------------
__global__ __launch_bounds__(256) void cvt_transpose_bf16(const float* __restrict__ src,
                                                          __bf16* __restrict__ dst,
                                                          int K, int N) {
  __shared__ __bf16 tile[64][65];
  const int tid = threadIdx.x;
  const int kb = blockIdx.y * 64, nb = blockIdx.x * 64;
#pragma unroll
  for (int i = 0; i < 16; ++i) {
    int s = tid + i * 256;
    int r = s >> 6, c = s & 63;
    tile[r][c] = (__bf16)src[(size_t)(kb + r) * N + nb + c];
  }
  __syncthreads();
#pragma unroll
  for (int i = 0; i < 16; ++i) {
    int s = tid + i * 256;
    int r = s >> 6, c = s & 63;
    dst[(size_t)(nb + r) * K + kb + c] = tile[c][r];
  }
}

__global__ __launch_bounds__(256) void zero_f32(float* __restrict__ p, int n) {
  int i = blockIdx.x * blockDim.x + threadIdx.x;
  if (i < n) p[i] = 0.0f;
}

// ---------------- GEMM1: h = GELU(x @ W1 + b1), bf16 out ----------------
__global__ __launch_bounds__(256) void gemm1_gelu(const __bf16* __restrict__ Xb,   // [NTOK,CDIM]
                                                  const __bf16* __restrict__ W1t,  // [HDIM,CDIM] (n-major)
                                                  const float* __restrict__ b1,    // [HDIM]
                                                  __bf16* __restrict__ Hb) {       // [NTOK,HDIM]
  __shared__ __align__(16) __bf16 smem[2 * BUF_ELT];   // double-buffered A+B tiles
  const int tid  = threadIdx.x;
  const int lane = tid & 31;
  const int wid  = tid >> 5;
  const int wm   = wid & 3;       // 4 wave rows * 32
  const int wn   = wid >> 2;      // 2 wave cols * 64
  const int mbase = blockIdx.y * TM;
  const int nbase = blockIdx.x * TN;
  const int l15  = lane & 15;
  const int khalf = lane >> 4;

  const __bf16* Ag = Xb  + (size_t)mbase * CDIM;
  const __bf16* Bg = W1t + (size_t)nbase * CDIM;

  const v8f z8 = {0.f,0.f,0.f,0.f,0.f,0.f,0.f,0.f};
  v8f acc[2][4];
#pragma unroll
  for (int im = 0; im < 2; ++im)
#pragma unroll
    for (int in = 0; in < 4; ++in) acc[im][in] = z8;

  const int NK = CDIM / TK;
  // prologue: stage tile 0 into buffer 0
  stage_tile(Ag, CDIM, smem, tid);
  stage_tile(Bg, CDIM, smem + TILE_ELT, tid);

  for (int k = 0; k < NK; ++k) {
    async_wait0();        // tile-k copies (issued one iteration ago) have landed
    __syncthreads();      // visible to all waves; prior reads of the other buffer done
    if (k + 1 < NK) {     // overlap: stage tile k+1 into the other buffer during compute
      __bf16* nb_ = smem + ((k + 1) & 1) * BUF_ELT;
      stage_tile(Ag + (size_t)(k + 1) * TK, CDIM, nb_, tid);
      stage_tile(Bg + (size_t)(k + 1) * TK, CDIM, nb_ + TILE_ELT, tid);
    }
    const __bf16* As = smem + (k & 1) * BUF_ELT;
    const __bf16* Bs = As + TILE_ELT;
#pragma unroll
    for (int ks = 0; ks < TK; ks += 32) {
      v16bf a[2], b[4];
#pragma unroll
      for (int im = 0; im < 2; ++im)
        a[im] = frag16(As + (wm * 32 + im * 16 + l15) * LDT + ks + khalf * 8, 16);
#pragma unroll
      for (int in = 0; in < 4; ++in)
        b[in] = frag16(Bs + (wn * 64 + in * 16 + l15) * LDT + ks + khalf * 8, 16);
#pragma unroll
      for (int im = 0; im < 2; ++im)
#pragma unroll
        for (int in = 0; in < 4; ++in)
          acc[im][in] = __builtin_amdgcn_wmma_f32_16x16x32_bf16(
              false, a[im], false, b[in], (short)0, acc[im][in], false, false);
    }
  }

  const int rh = lane >> 4;
#pragma unroll
  for (int im = 0; im < 2; ++im)
#pragma unroll
    for (int in = 0; in < 4; ++in) {
      int col = nbase + wn * 64 + in * 16 + l15;
      float bias = b1[col];
      int rbase = mbase + wm * 32 + im * 16 + rh * 8;
#pragma unroll
      for (int v = 0; v < 8; ++v) {
        float q = acc[im][in][v] + bias;
        float g = 0.5f * q * (1.0f + erff(q * 0.70710678118654752f));
        Hb[(size_t)(rbase + v) * HDIM + col] = (__bf16)g;
      }
    }
}

// -------- GEMM2: query = h @ W2 + b2, fused scores = query @ tq^T (atomic) --------
__global__ __launch_bounds__(256) void gemm2_scores(const __bf16* __restrict__ Hb,   // [NTOK,HDIM]
                                                    const __bf16* __restrict__ W2t,  // [CDIM,HDIM] (n-major)
                                                    const float* __restrict__ b2,    // [CDIM]
                                                    const float* __restrict__ TQ,    // [EXP,CDIM]
                                                    float* __restrict__ scores) {    // [NTOK,EXP]
  __shared__ __align__(16) __bf16 smem[2 * BUF_ELT];
  const int tid  = threadIdx.x;
  const int lane = tid & 31;
  const int wid  = tid >> 5;
  const int wm   = wid & 3;
  const int wn   = wid >> 2;
  const int mbase = blockIdx.y * TM;
  const int nbase = blockIdx.x * TN;
  const int l15  = lane & 15;
  const int khalf = lane >> 4;

  const __bf16* Ag = Hb  + (size_t)mbase * HDIM;
  const __bf16* Bg = W2t + (size_t)nbase * HDIM;

  const v8f z8 = {0.f,0.f,0.f,0.f,0.f,0.f,0.f,0.f};
  v8f acc[2][4];
#pragma unroll
  for (int im = 0; im < 2; ++im)
#pragma unroll
    for (int in = 0; in < 4; ++in) acc[im][in] = z8;

  const int NK = HDIM / TK;
  stage_tile(Ag, HDIM, smem, tid);
  stage_tile(Bg, HDIM, smem + TILE_ELT, tid);

  for (int k = 0; k < NK; ++k) {
    async_wait0();
    __syncthreads();
    if (k + 1 < NK) {
      __bf16* nb_ = smem + ((k + 1) & 1) * BUF_ELT;
      stage_tile(Ag + (size_t)(k + 1) * TK, HDIM, nb_, tid);
      stage_tile(Bg + (size_t)(k + 1) * TK, HDIM, nb_ + TILE_ELT, tid);
    }
    const __bf16* As = smem + (k & 1) * BUF_ELT;
    const __bf16* Bs = As + TILE_ELT;
#pragma unroll
    for (int ks = 0; ks < TK; ks += 32) {
      v16bf a[2], b[4];
#pragma unroll
      for (int im = 0; im < 2; ++im)
        a[im] = frag16(As + (wm * 32 + im * 16 + l15) * LDT + ks + khalf * 8, 16);
#pragma unroll
      for (int in = 0; in < 4; ++in)
        b[in] = frag16(Bs + (wn * 64 + in * 16 + l15) * LDT + ks + khalf * 8, 16);
#pragma unroll
      for (int im = 0; im < 2; ++im)
#pragma unroll
        for (int in = 0; in < 4; ++in)
          acc[im][in] = __builtin_amdgcn_wmma_f32_16x16x32_bf16(
              false, a[im], false, b[in], (short)0, acc[im][in], false, false);
    }
  }

  // epilogue: +b2, contract against type_queries across this wave's 4 col-tiles,
  // width-16 shuffle reduce, atomic-accumulate per (row, expert)
  const int rh = lane >> 4;
#pragma unroll
  for (int im = 0; im < 2; ++im) {
    int rbase = mbase + wm * 32 + im * 16 + rh * 8;
    float qv[4][8];
#pragma unroll
    for (int in = 0; in < 4; ++in) {
      int col = nbase + wn * 64 + in * 16 + l15;
      float bias = b2[col];
#pragma unroll
      for (int v = 0; v < 8; ++v) qv[in][v] = acc[im][in][v] + bias;
    }
#pragma unroll
    for (int e = 0; e < EXP; ++e) {
      float tqv[4];
#pragma unroll
      for (int in = 0; in < 4; ++in)
        tqv[in] = TQ[(size_t)e * CDIM + nbase + wn * 64 + in * 16 + l15];
#pragma unroll
      for (int v = 0; v < 8; ++v) {
        float r = qv[0][v] * tqv[0] + qv[1][v] * tqv[1] + qv[2][v] * tqv[2] + qv[3][v] * tqv[3];
        r += __shfl_xor(r, 1, 16);
        r += __shfl_xor(r, 2, 16);
        r += __shfl_xor(r, 4, 16);
        r += __shfl_xor(r, 8, 16);
        if (l15 == 0) atomicAdd(&scores[(size_t)(rbase + v) * EXP + e], r);
      }
    }
  }
}

// ---------------- finalize: gate, noisy, top-2, masked softmax ----------------
__global__ __launch_bounds__(256) void router_finalize(const float* __restrict__ X,
                                                       const float* __restrict__ noise,
                                                       const float* __restrict__ scores,
                                                       const float* __restrict__ noise_w,
                                                       const float* __restrict__ noise_b,
                                                       const float* __restrict__ temp_p,
                                                       float* __restrict__ out_router,
                                                       float* __restrict__ out_idx) {
  const int lane = threadIdx.x & 31;
  const int wid  = threadIdx.x >> 5;
  const int r = blockIdx.x * 8 + wid;   // one wave per token row
  if (r >= NTOK) return;

  const float* xr = X + (size_t)r * CDIM;
  float s = 0.0f;
#pragma unroll 8
  for (int i = 0; i < CDIM / 32; ++i) {
    int c = i * 32 + lane;
    s += xr[c] * noise_w[c];
  }
  s += __shfl_xor(s, 16, 32);
  s += __shfl_xor(s, 8, 32);
  s += __shfl_xor(s, 4, 32);
  s += __shfl_xor(s, 2, 32);
  s += __shfl_xor(s, 1, 32);

  if (lane == 0) {
    float temp = temp_p[0];
    float gate = 1.0f / (1.0f + expf(-(s + noise_b[0])));
    float ny[EXP];
#pragma unroll
    for (int e = 0; e < EXP; ++e)
      ny[e] = scores[(size_t)r * EXP + e] + temp * noise[(size_t)r * EXP + e] * gate;
    int i1 = 0;
#pragma unroll
    for (int e = 1; e < EXP; ++e) if (ny[e] > ny[i1]) i1 = e;
    int i2 = (i1 == 0) ? 1 : 0;
#pragma unroll
    for (int e = 0; e < EXP; ++e) {
      if (e == i1 || e == i2) continue;
      if (ny[e] > ny[i2]) i2 = e;
    }
    float tau = temp + 1e-6f;
    float e1 = expf((ny[i2] - ny[i1]) / tau);
    float inv = 1.0f / (1.0f + e1);
#pragma unroll
    for (int e = 0; e < EXP; ++e)
      out_router[(size_t)r * EXP + e] = (e == i1) ? inv : ((e == i2) ? e1 * inv : 0.0f);
    out_idx[(size_t)r * 2 + 0] = (float)i1;
    out_idx[(size_t)r * 2 + 1] = (float)i2;
  }
}

extern "C" void kernel_launch(void* const* d_in, const int* in_sizes, int n_in,
                              void* d_out, int out_size, void* d_ws, size_t ws_size,
                              hipStream_t stream) {
  const float* x     = (const float*)d_in[0];
  const float* noise = (const float*)d_in[1];
  const float* W1    = (const float*)d_in[2];
  const float* b1    = (const float*)d_in[3];
  const float* W2    = (const float*)d_in[4];
  const float* b2    = (const float*)d_in[5];
  const float* tq    = (const float*)d_in[6];
  const float* nw    = (const float*)d_in[7];
  const float* nb    = (const float*)d_in[8];
  const float* temp  = (const float*)d_in[9];

  // workspace layout
  char* ws = (char*)d_ws;
  __bf16* Xb    = (__bf16*)ws;                               //  64 MB  [NTOK,CDIM]
  __bf16* W1t   = (__bf16*)(ws + (size_t)(64u << 20));       //   8 MB  [HDIM,CDIM]
  __bf16* W2t   = (__bf16*)(ws + (size_t)(72u << 20));       //   8 MB  [CDIM,HDIM]
  float*  scr   = (float*) (ws + (size_t)(80u << 20));       //   1 MB  [NTOK,EXP]
  __bf16* Hb    = (__bf16*)(ws + (size_t)(81u << 20));       // 256 MB  [NTOK,HDIM]

  // 1) x -> bf16; weights -> transposed bf16 (n-major, so GEMM staging is contiguous)
  cvt_f32_bf16<<<NTOK * CDIM / 4 / 256, 256, 0, stream>>>(x, Xb, NTOK * CDIM / 4);
  cvt_transpose_bf16<<<dim3(HDIM / 64, CDIM / 64), 256, 0, stream>>>(W1, W1t, CDIM, HDIM);
  cvt_transpose_bf16<<<dim3(CDIM / 64, HDIM / 64), 256, 0, stream>>>(W2, W2t, HDIM, CDIM);
  // 2) zero score accumulator
  zero_f32<<<(NTOK * EXP + 255) / 256, 256, 0, stream>>>(scr, NTOK * EXP);
  // 3) h = GELU(x @ W1 + b1)
  gemm1_gelu<<<dim3(HDIM / TN, NTOK / TM), 256, 0, stream>>>(Xb, W1t, b1, Hb);
  // 4) query = h @ W2 + b2, fused scores accumulation
  gemm2_scores<<<dim3(CDIM / TN, NTOK / TM), 256, 0, stream>>>(Hb, W2t, b2, tq, scr);
  // 5) gate + noisy + top-2 + masked softmax
  float* out_router = (float*)d_out;
  float* out_idx    = (float*)d_out + (size_t)NTOK * EXP;
  router_finalize<<<NTOK / 8, 256, 0, stream>>>(x, noise, scr, nw, nb, temp,
                                                out_router, out_idx);
}
